// RegClsModuleX_16819091931542
// MI455X (gfx1250) — matrix operand order repaired and checked by reading, not compiled, not run on hardware
//
#include <hip/hip_runtime.h>
#include <math.h>

// ---------------------------------------------------------------------------
// Types for CDNA5 WMMA
// ---------------------------------------------------------------------------
typedef __attribute__((ext_vector_type(16))) __bf16 v16bf;
typedef __attribute__((ext_vector_type(8)))  float  v8f;

union FragBF16 {
    v16bf v;
    unsigned int u[8];
};

__device__ __forceinline__ unsigned short f2bf(float f) {
    union { float f; unsigned int u; } c;
    c.f = f;
    unsigned int u = c.u;
    unsigned int r = (u + 0x7FFFu + ((u >> 16) & 1u)) >> 16;  // RNE
    return (unsigned short)r;
}

// ---------------------------------------------------------------------------
// Weight repack: f32 OIHW [co][ci*9+tap] -> bf16 tap-major [co][tap*256+ci]
// ---------------------------------------------------------------------------
__global__ __launch_bounds__(256) void repack_w_kernel(
    const float* __restrict__ src, unsigned short* __restrict__ dst,
    int total) {
    int i = blockIdx.x * 256 + threadIdx.x;
    if (i >= total) return;
    int matco = i / 2304;      // (mat*256 + co)
    int kp    = i - matco * 2304;
    int tap   = kp >> 8;       // 0..8
    int ci    = kp & 255;
    dst[i] = f2bf(src[(size_t)matco * 2304 + ci * 9 + tap]);
}

// ---------------------------------------------------------------------------
// Activation repack: NCHW f32 -> NHWC bf16 (C = 256)
// ---------------------------------------------------------------------------
__global__ __launch_bounds__(256) void nchw_to_hwc_bf16(
    const float* __restrict__ src, unsigned short* __restrict__ dst,
    int Bsz, int HW) {
    size_t i = (size_t)blockIdx.x * 256 + threadIdx.x;
    size_t total = (size_t)Bsz * 256 * HW;
    if (i >= total) return;
    int c  = (int)(i & 255);
    int t  = (int)(i >> 8);    // b*HW + pix
    int b  = t / HW;
    int pix = t - b * HW;
    dst[i] = f2bf(src[((size_t)b * 256 + c) * HW + pix]);
}

// ---------------------------------------------------------------------------
// Implicit-GEMM 3x3 conv (Cin=Cout=256, K=2304 tap-major) + BN(+bias) + ReLU.
// NHWC bf16 input. 256x128 tile / 256-thread block; 8 waves x (4x4) WMMA
// 16x16x32 bf16 frags (16 WMMA per wave per K-step, 1.0 b128-LDS-read/WMMA).
// Double-buffered LDS, software prefetch, 1 barrier per K-step.
// ---------------------------------------------------------------------------
#define LSTR 40                 // LDS row stride in bf16 (80B, 16B-aligned)
#define ATILE_SH (256 * LSTR)   // A tile: 256 rows
#define BTILE_SH (128 * LSTR)   // B tile: 128 rows

template <int PAD>
__global__ __launch_bounds__(256) void conv_gemm_wmma(
    const unsigned short* __restrict__ xhwc,  // (Bsz,H,W,256) bf16
    const unsigned short* __restrict__ wT,    // (256,2304) bf16 tap-major
    const float* __restrict__ gamma, const float* __restrict__ beta,
    const float* __restrict__ mean, const float* __restrict__ var,
    const float* __restrict__ cbias,          // nullable conv bias
    float* __restrict__ out,                  // (Bsz,256,Ho,Wo) f32
    int Bsz, int H, int W, int Ho, int Wo) {
    __shared__ unsigned short As[2 * ATILE_SH];
    __shared__ unsigned short Bs[2 * BTILE_SH];

    const int tid    = threadIdx.x;
    const int lane   = tid & 31;
    const int hlane  = lane & 15;
    const bool hiHalf = lane >= 16;
    const int wid = tid >> 5;
    const int wm  = wid & 3;   // wave row block (64 rows)
    const int wn  = wid >> 2;  // wave col block (64 cols)
    const int m0  = blockIdx.x * 256;
    const int n0  = blockIdx.y * 128;
    const int HoWo = Ho * Wo;
    const int M    = Bsz * HoWo;

    // ---- A staging map: thread -> one pixel row, full 32-channel K-strip
    const int am = m0 + tid;
    const bool avalid = am < M;
    int ab = 0, aoh = 0, aow = 0;
    if (avalid) {
        ab = am / HoWo;
        int rr = am - ab * HoWo;
        aoh = rr / Wo;
        aow = rr - aoh * Wo;
    }
    const int aslotOfs = tid * LSTR;

    // ---- B staging map: 2 threads per row, 16 contiguous K each
    const int bln = tid >> 1;
    const int bk0 = (tid & 1) * 16;
    const unsigned short* bsrc = wT + (size_t)(n0 + bln) * 2304 + bk0;
    const int bslotOfs = bln * LSTR + bk0;

    // ---- fragment K-column tables (CDNA5 16-bit A 16x32 / B 32x16 layouts)
    int colA[8], colB[8];
#pragma unroll
    for (int j = 0; j < 8; ++j) {
        int lo = (j < 4) ? 2 * j : 16 + 2 * (j - 4);
        colA[j] = hiHalf ? lo + 8 : lo;
        colB[j] = hiHalf ? lo + 8 : lo;
    }

    // ---- fused BN(+bias) epilogue params
    float scl[4], sft[4];
#pragma unroll
    for (int fj = 0; fj < 4; ++fj) {
        int co = n0 + wn * 64 + fj * 16 + hlane;
        float sc = gamma[co] * rsqrtf(var[co] + 1e-5f);
        float cb = cbias ? cbias[co] : 0.0f;
        scl[fj] = sc;
        sft[fj] = (cb - mean[co]) * sc + beta[co];
    }

    v8f acc[4][4];
#pragma unroll
    for (int i = 0; i < 4; ++i)
#pragma unroll
        for (int j = 0; j < 4; ++j)
#pragma unroll
            for (int e = 0; e < 8; ++e) acc[i][j][e] = 0.0f;

    // ---- tile fetch (K-step `step` in 0..71, k = step*32, tap = step>>3)
    auto fetch = [&](int step, uint4* a4, uint4& b0, uint4& b1) {
        int tap = step >> 3;
        int cb  = step & 7;
        int kr  = tap / 3;
        int ks  = tap - 3 * kr;
        int ih  = aoh + kr - PAD;
        int iw  = aow + ks - PAD;
        bool ok = avalid;
        if (PAD) ok = ok && (ih >= 0) && (ih < H) && (iw >= 0) && (iw < W);
        size_t abase =
            ok ? ((size_t)((ab * H + ih) * W + iw) * 256 + cb * 32) : 0;
        const uint4* ap = (const uint4*)(xhwc + abase);
#pragma unroll
        for (int q = 0; q < 4; ++q) a4[q] = ap[q];
        if (!ok) {
#pragma unroll
            for (int q = 0; q < 4; ++q)
                a4[q].x = a4[q].y = a4[q].z = a4[q].w = 0u;
        }
        const uint4* bp = (const uint4*)(bsrc + step * 32);
        b0 = bp[0];
        b1 = bp[1];
    };

    uint4 fa[4], fb0, fb1;
    fetch(0, fa, fb0, fb1);

#pragma unroll 2
    for (int step = 0; step < 72; ++step) {
        const int p = step & 1;
        unsigned short* abuf = As + p * ATILE_SH;
        unsigned short* bbuf = Bs + p * BTILE_SH;

        // ---- store prefetched tile into this step's LDS buffer
#pragma unroll
        for (int q = 0; q < 4; ++q) ((uint4*)(abuf + aslotOfs))[q] = fa[q];
        ((uint4*)(bbuf + bslotOfs))[0] = fb0;
        ((uint4*)(bbuf + bslotOfs))[1] = fb1;

        // ---- prefetch next tile (overlaps with WMMA below)
        int nstep = (step + 1 < 72) ? step + 1 : 71;
        fetch(nstep, fa, fb0, fb1);

        __syncthreads();

        // ---- gather fragments from LDS
        FragBF16 a[4], b[4];
#pragma unroll
        for (int fi = 0; fi < 4; ++fi) {
            const unsigned short* ap2 =
                abuf + (wm * 64 + fi * 16 + hlane) * LSTR;
#pragma unroll
            for (int j = 0; j < 8; ++j)
                a[fi].u[j] = *(const unsigned int*)(ap2 + colA[j]);
        }
#pragma unroll
        for (int fj = 0; fj < 4; ++fj) {
            const unsigned short* bp2 =
                bbuf + (wn * 64 + fj * 16 + hlane) * LSTR;
#pragma unroll
            for (int j = 0; j < 8; ++j)
                b[fj].u[j] = *(const unsigned int*)(bp2 + colB[j]);
        }

        // ---- 16 WMMAs per wave per K-step
#pragma unroll
        for (int fi = 0; fi < 4; ++fi)
#pragma unroll
            for (int fj = 0; fj < 4; ++fj)
                acc[fi][fj] = __builtin_amdgcn_wmma_f32_16x16x32_bf16(
                    false, a[fi].v, false, b[fj].v, (short)0, acc[fi][fj],
                    false, false);
    }

    // ---- epilogue: BN + ReLU, NCHW scatter
#pragma unroll
    for (int fi = 0; fi < 4; ++fi) {
#pragma unroll
        for (int v = 0; v < 8; ++v) {
            int m = m0 + wm * 64 + fi * 16 + (hiHalf ? 8 + v : v);
            if (m < M) {
                int b  = m / HoWo;
                int rr = m - b * HoWo;
                size_t ob = (size_t)b * 256 * HoWo + rr;
#pragma unroll
                for (int fj = 0; fj < 4; ++fj) {
                    int co = n0 + wn * 64 + fj * 16 + hlane;
                    float r = acc[fi][fj][v] * scl[fj] + sft[fj];
                    out[ob + (size_t)co * HoWo] = fmaxf(r, 0.0f);
                }
            }
        }
    }
}

// ---------------------------------------------------------------------------
// Depthwise xcorr (29x29 (*) 5x5 -> 25x25) with on-device softmax level
// weight, accumulated across pyramid levels. One block per (b,c) plane.
// ---------------------------------------------------------------------------
__global__ __launch_bounds__(128) void xcorr_acc_kernel(
    const float* __restrict__ xs,   // (B*C,29,29)
    const float* __restrict__ zs,   // (B*C,5,5)
    float* __restrict__ out,        // (B*C,25,25)
    const float* __restrict__ wraw, // (3,) raw level logits
    int level, int init) {
    __shared__ float sx[841];
    __shared__ float sz[25];
    int p = blockIdx.x;
    const float* xp = xs + (size_t)p * 841;
    const float* zp = zs + (size_t)p * 25;
    for (int i = threadIdx.x; i < 841; i += 128) sx[i] = xp[i];
    if (threadIdx.x < 25) sz[threadIdx.x] = zp[threadIdx.x];
    __syncthreads();

    float w0 = wraw[0], w1 = wraw[1], w2 = wraw[2];
    float mx = fmaxf(w0, fmaxf(w1, w2));
    float e0 = expf(w0 - mx), e1 = expf(w1 - mx), e2 = expf(w2 - mx);
    float wl = ((level == 0) ? e0 : (level == 1) ? e1 : e2) / (e0 + e1 + e2);

    for (int o = threadIdx.x; o < 625; o += 128) {
        int oy = o / 25, ox = o - 25 * oy;
        float acc = 0.0f;
#pragma unroll
        for (int r = 0; r < 5; ++r)
#pragma unroll
            for (int s = 0; s < 5; ++s)
                acc += sx[(oy + r) * 29 + ox + s] * sz[r * 5 + s];
        float v = acc * wl;
        size_t oi = (size_t)p * 625 + o;
        if (init) out[oi] = v;
        else      out[oi] += v;
    }
}

// ---------------------------------------------------------------------------
// Prediction heads (tiny): 3x3 SAME convs, 256 -> 1 and 256 -> 4.
// ---------------------------------------------------------------------------
__global__ __launch_bounds__(256) void cls_pred_kernel(
    const float* __restrict__ feat, const float* __restrict__ w,
    const float* __restrict__ b0, float* __restrict__ out) {
    int idx = blockIdx.x * 256 + threadIdx.x;
    if (idx >= 32 * 625) return;
    int b = idx / 625, o = idx - b * 625;
    int oy = o / 25, ox = o - 25 * oy;
    const float* fb = feat + (size_t)b * 256 * 625;
    float acc = b0[0];
    for (int ci = 0; ci < 256; ++ci) {
        const float* fp = fb + ci * 625;
        const float* wp = w + ci * 9;
#pragma unroll
        for (int r = 0; r < 3; ++r) {
            int iy = oy + r - 1;
            if (iy < 0 || iy >= 25) continue;
#pragma unroll
            for (int s = 0; s < 3; ++s) {
                int ix = ox + s - 1;
                if (ix < 0 || ix >= 25) continue;
                acc += fp[iy * 25 + ix] * wp[r * 3 + s];
            }
        }
    }
    out[idx] = 0.1f * acc;
}

__global__ __launch_bounds__(256) void reg_pred_kernel(
    const float* __restrict__ feat, const float* __restrict__ w,
    const float* __restrict__ bvec, const float* __restrict__ adjust,
    const float* __restrict__ bias4, float* __restrict__ out) {
    int idx = blockIdx.x * 256 + threadIdx.x;
    if (idx >= 32 * 4 * 625) return;
    int b = idx / 2500;
    int rem = idx - b * 2500;
    int ch = rem / 625, o = rem - ch * 625;
    int oy = o / 25, ox = o - 25 * oy;
    const float* fb = feat + (size_t)b * 256 * 625;
    const float* wc = w + (size_t)ch * 2304;
    float acc = bvec[ch];
    for (int ci = 0; ci < 256; ++ci) {
        const float* fp = fb + ci * 625;
        const float* wp = wc + ci * 9;
#pragma unroll
        for (int r = 0; r < 3; ++r) {
            int iy = oy + r - 1;
            if (iy < 0 || iy >= 25) continue;
#pragma unroll
            for (int s = 0; s < 3; ++s) {
                int ix = ox + s - 1;
                if (ix < 0 || ix >= 25) continue;
                acc += fp[iy * 25 + ix] * wp[r * 3 + s];
            }
        }
    }
    out[idx] = expf(adjust[0] * acc + bias4[ch]);
}

// ---------------------------------------------------------------------------
// Launcher
// ---------------------------------------------------------------------------
extern "C" void kernel_launch(void* const* d_in, const int* in_sizes, int n_in,
                              void* d_out, int out_size, void* d_ws,
                              size_t ws_size, hipStream_t stream) {
    (void)in_sizes; (void)n_in; (void)out_size; (void)ws_size;
    const float* x_feats      = (const float*)d_in[0];
    const float* z_feats      = (const float*)d_in[1];
    const float* enc_w        = (const float*)d_in[2];
    const float* enc_gamma    = (const float*)d_in[3];
    const float* enc_beta     = (const float*)d_in[4];
    const float* enc_mean     = (const float*)d_in[5];
    const float* enc_var      = (const float*)d_in[6];
    const float* cls_weight   = (const float*)d_in[7];
    const float* reg_weight   = (const float*)d_in[8];
    const float* cls_tower_w  = (const float*)d_in[9];
    const float* cls_tower_b  = (const float*)d_in[10];
    const float* cls_bn_gamma = (const float*)d_in[11];
    const float* cls_bn_beta  = (const float*)d_in[12];
    const float* cls_bn_mean  = (const float*)d_in[13];
    const float* cls_bn_var   = (const float*)d_in[14];
    const float* reg_tower_w  = (const float*)d_in[15];
    const float* reg_tower_b  = (const float*)d_in[16];
    const float* reg_bn_gamma = (const float*)d_in[17];
    const float* reg_bn_beta  = (const float*)d_in[18];
    const float* reg_bn_mean  = (const float*)d_in[19];
    const float* reg_bn_var   = (const float*)d_in[20];
    const float* cls_pred_w   = (const float*)d_in[21];
    const float* cls_pred_b   = (const float*)d_in[22];
    const float* bbox_pred_w  = (const float*)d_in[23];
    const float* bbox_pred_b  = (const float*)d_in[24];
    const float* adjust       = (const float*)d_in[25];
    const float* bias4        = (const float*)d_in[26];

    char* ws = (char*)d_ws;
    size_t off = 0;
    auto alloc = [&](size_t bytes) -> char* {
        char* p = ws + off;
        off = (off + bytes + 255) & ~(size_t)255;
        return p;
    };
    const size_t ENC_W_ELEMS   = 3ull * 4 * 256 * 256 * 9;  // 7,077,888
    const size_t TOWER_W_ELEMS = 256ull * 256 * 9;          // 589,824
    const size_t XL = 32ull * 256 * 961;  // one level of x_feats
    const size_t ZL = 32ull * 256 * 49;   // one level of z_feats
    unsigned short* encw_bf = (unsigned short*)alloc(ENC_W_ELEMS * 2);
    unsigned short* clsw_bf = (unsigned short*)alloc(TOWER_W_ELEMS * 2);
    unsigned short* regw_bf = (unsigned short*)alloc(TOWER_W_ELEMS * 2);
    unsigned short* xhwc    = (unsigned short*)alloc(XL * 2);
    unsigned short* zhwc    = (unsigned short*)alloc(ZL * 2);
    unsigned short* fhwc    = (unsigned short*)alloc(32ull * 256 * 625 * 2);
    float* bufX    = (float*)alloc(32ull * 256 * 29 * 29 * 4);
    float* bufZ    = (float*)alloc(32ull * 256 * 5 * 5 * 4);
    float* clsFeat = (float*)alloc(32ull * 256 * 625 * 4);
    float* regFeat = (float*)alloc(32ull * 256 * 625 * 4);

    // 1) repack weights to bf16, K reordered tap-major
    repack_w_kernel<<<(ENC_W_ELEMS + 255) / 256, 256, 0, stream>>>(
        enc_w, encw_bf, (int)ENC_W_ELEMS);
    repack_w_kernel<<<(TOWER_W_ELEMS + 255) / 256, 256, 0, stream>>>(
        cls_tower_w, clsw_bf, (int)TOWER_W_ELEMS);
    repack_w_kernel<<<(TOWER_W_ELEMS + 255) / 256, 256, 0, stream>>>(
        reg_tower_w, regw_bf, (int)TOWER_W_ELEMS);

    // 2) per-level: repack activations NHWC-bf16 (shared by both branches),
    //    run encoders + depthwise xcorr accumulate
    const int MX = 32 * 29 * 29;  // 26912
    const int MZ = 32 * 5 * 5;    // 800
    dim3 gx((MX + 255) / 256, 2), gz((MZ + 255) / 256, 2);
    for (int i = 0; i < 3; ++i) {
        nchw_to_hwc_bf16<<<(unsigned)((XL + 255) / 256), 256, 0, stream>>>(
            x_feats + XL * i, xhwc, 32, 961);
        nchw_to_hwc_bf16<<<(unsigned)((ZL + 255) / 256), 256, 0, stream>>>(
            z_feats + ZL * i, zhwc, 32, 49);
        for (int br = 0; br < 2; ++br) {
            int es = i * 4 + ((br == 0) ? 0 : 2), ek = es + 1;
            float* feat = (br == 0) ? clsFeat : regFeat;
            const float* wvec = (br == 0) ? cls_weight : reg_weight;
            conv_gemm_wmma<0><<<gx, 256, 0, stream>>>(
                xhwc, encw_bf + (size_t)es * TOWER_W_ELEMS,
                enc_gamma + es * 256, enc_beta + es * 256,
                enc_mean + es * 256, enc_var + es * 256, nullptr, bufX,
                32, 31, 31, 29, 29);
            conv_gemm_wmma<0><<<gz, 256, 0, stream>>>(
                zhwc, encw_bf + (size_t)ek * TOWER_W_ELEMS,
                enc_gamma + ek * 256, enc_beta + ek * 256,
                enc_mean + ek * 256, enc_var + ek * 256, nullptr, bufZ,
                32, 7, 7, 5, 5);
            xcorr_acc_kernel<<<32 * 256, 128, 0, stream>>>(
                bufX, bufZ, feat, wvec, i, (i == 0) ? 1 : 0);
        }
    }

    // 3) towers (pad=1 SAME) straight into d_out
    float* outF = (float*)d_out;
    float* cls_feat_out = outF + 100000;   // after cls_pred(20k)+reg_pred(80k)
    float* reg_feat_out = outF + 5220000;  // after cls_feat (5,120,000)
    const size_t FT = 32ull * 256 * 625;
    const int MT = 32 * 625;
    dim3 gt((MT + 255) / 256, 2);
    nchw_to_hwc_bf16<<<(unsigned)((FT + 255) / 256), 256, 0, stream>>>(
        clsFeat, fhwc, 32, 625);
    conv_gemm_wmma<1><<<gt, 256, 0, stream>>>(
        fhwc, clsw_bf, cls_bn_gamma, cls_bn_beta, cls_bn_mean, cls_bn_var,
        cls_tower_b, cls_feat_out, 32, 25, 25, 25, 25);
    nchw_to_hwc_bf16<<<(unsigned)((FT + 255) / 256), 256, 0, stream>>>(
        regFeat, fhwc, 32, 625);
    conv_gemm_wmma<1><<<gt, 256, 0, stream>>>(
        fhwc, regw_bf, reg_bn_gamma, reg_bn_beta, reg_bn_mean, reg_bn_var,
        reg_tower_b, reg_feat_out, 32, 25, 25, 25, 25);

    // 4) prediction heads
    cls_pred_kernel<<<(32 * 625 + 255) / 256, 256, 0, stream>>>(
        cls_feat_out, cls_pred_w, cls_pred_b, outF);
    reg_pred_kernel<<<(32 * 4 * 625 + 255) / 256, 256, 0, stream>>>(
        reg_feat_out, bbox_pred_w, bbox_pred_b, adjust, bias4, outF + 20000);
}
